// Model_88098369176363
// MI455X (gfx1250) — matrix operand order repaired and checked by reading, not compiled, not run on hardware
//
#include <hip/hip_runtime.h>
#include <hip/hip_bf16.h>

// ---------------------------------------------------------------------------
// Shapes
// ---------------------------------------------------------------------------
#define VV   50257
#define EE   300
#define EPAD 320
#define HH   512
#define BB   4
#define TT   512
#define G4H  2048               // 4*H
#define MROWS 2048              // B*T
#define NLIN_PAD 50688          // pad V up to grid coverage (99*512)

typedef __attribute__((ext_vector_type(16))) __bf16 v16bf;
typedef __attribute__((ext_vector_type(8)))  __bf16 v8bf;
typedef __attribute__((ext_vector_type(8)))  float  v8f;

// ---------------------------------------------------------------------------
// WMMA helpers (wave32, 16x16x32 bf16 -> f32)
// A operand layout (ISA 7.12.2, 16-bit A 16x32): lane m(0..15) row m holds
// K=k0..k0+7 and k0+16..k0+23 ; lane m+16 holds K=k0+8..15 and k0+24..31.
// B operand (columns of B == rows of B^T): lane n(0..15) col n holds
// K=k0..k0+7 and k0+16..k0+23 ; lane n+16 holds K=k0+8..15 and k0+24..31 (same as A).
// ---------------------------------------------------------------------------
__device__ __forceinline__ v16bf frag_A(const __bf16* A, int ldk, int k0, int lane) {
    int row = lane & 15;
    int hi  = lane >> 4;
    const __bf16* p = A + (size_t)row * ldk + k0 + hi * 8;
    v8bf lo = *(const v8bf*)p;
    v8bf h8 = *(const v8bf*)(p + 16);
    v16bf r;
#pragma unroll
    for (int i = 0; i < 8; ++i) { r[i] = lo[i]; r[i + 8] = h8[i]; }
    return r;
}

__device__ __forceinline__ v16bf frag_B(const __bf16* BT, int ldk, int n0, int k0, int lane) {
    int col = lane & 15;
    int hi  = lane >> 4;
    const __bf16* p = BT + (size_t)(n0 + col) * ldk + k0 + hi * 8;
    v8bf lo = *(const v8bf*)p;
    v8bf h8 = *(const v8bf*)(p + 16);
    v16bf r;
#pragma unroll
    for (int i = 0; i < 8; ++i) { r[i] = lo[i]; r[i + 8] = h8[i]; }
    return r;
}

__device__ __forceinline__ v8f wmma_bf16(v16bf a, v16bf b, v8f c) {
    return __builtin_amdgcn_wmma_f32_16x16x32_bf16(false, a, false, b, (short)0, c, false, false);
}

// ---------------------------------------------------------------------------
// Embedding gather: tokens[B*T] -> Aemb bf16 [2048][320], cols 300..319 zero
// ---------------------------------------------------------------------------
__global__ void gather_emb_kernel(const int* __restrict__ tokens,
                                  const float* __restrict__ emb,
                                  __bf16* __restrict__ Aemb) {
    int r = blockIdx.x;                       // 0..2047 (= b*T + t)
    int tok = tokens[r];
    for (int e = threadIdx.x; e < EPAD; e += 256) {
        float v = (e < EE) ? emb[(size_t)tok * EE + e] : 0.f;
        Aemb[(size_t)r * EPAD + e] = (__bf16)v;
    }
}

// ---------------------------------------------------------------------------
// f32 [K][N] -> bf16 transposed+padded [Np][Kp]; 16x16 LDS tile transpose
// grid = (Kp/16, Np/16), block = 256
// ---------------------------------------------------------------------------
__global__ void transpose_bf16_kernel(const float* __restrict__ src,
                                      __bf16* __restrict__ dst,
                                      int K, int N, int Kp, int Np) {
    __shared__ float tile[16][17];
    int tx = threadIdx.x & 15, ty = threadIdx.x >> 4;
    int k0 = blockIdx.x * 16, n0 = blockIdx.y * 16;
    int k = k0 + ty, n = n0 + tx;
    tile[ty][tx] = (k < K && n < N) ? src[(size_t)k * N + n] : 0.f;
    __syncthreads();
    int dn = n0 + ty, dk = k0 + tx;
    if (dn < Np && dk < Kp)
        dst[(size_t)dn * Kp + dk] = (__bf16)tile[tx][ty];
}

// ---------------------------------------------------------------------------
// Generic WMMA GEMM: C[M][Nstride] = A[M][K]bf16 @ BT[Npad][K]bf16 (+bias)
// block = 256 (8 waves). Block tile: 64 (M) x 512 (N); wave tile 64x64
// (4 M-subtiles x 4 N-subtiles -> 16 accumulators, 4x register reuse of B).
// grid = (ceil(N/512), M/64). K multiple of 32, K <= 512, M multiple of 64.
// ---------------------------------------------------------------------------
__global__ __launch_bounds__(256)
void gemm_bf16_kernel(const __bf16* __restrict__ A, const __bf16* __restrict__ BT,
                      float* __restrict__ C, const float* __restrict__ bias,
                      int N, int Nstride, int K) {
    __shared__ __align__(16) __bf16 As[64 * 512];   // 64 KB
    int tid   = threadIdx.x;
    int mBase = blockIdx.y * 64;
    int nBlk  = blockIdx.x * 512;

    // stage A tile [64][K] into LDS (coalesced 16B copies)
    {
        const uint4* s = (const uint4*)(A + (size_t)mBase * K);
        uint4* d = (uint4*)As;
        int n128 = 64 * K / 8;                      // 16-byte packets
        for (int i = tid; i < n128; i += 256) d[i] = s[i];
    }
    __syncthreads();

    int wave = tid >> 5, lane = tid & 31;
    int n0w = nBlk + wave * 64;
    v8f acc[4][4] = {};                              // [msub][nsub]
    for (int k0 = 0; k0 < K; k0 += 32) {
        // prefetch the k-tile two steps ahead of our 64 B rows (global_prefetch_b8 path)
        if (k0 + 64 < K) {
            const __bf16* pf = BT + (size_t)(n0w + lane) * K + k0 + 64;
            __builtin_prefetch(pf, 0, 0);
            __builtin_prefetch(pf + (size_t)32 * K, 0, 0);
        }
        v16bf a[4];
#pragma unroll
        for (int ms = 0; ms < 4; ++ms)
            a[ms] = frag_A(As + (size_t)ms * 16 * K, K, k0, lane);
#pragma unroll
        for (int j = 0; j < 4; ++j) {
            v16bf b = frag_B(BT, K, n0w + j * 16, k0, lane);
#pragma unroll
            for (int ms = 0; ms < 4; ++ms)
                acc[ms][j] = wmma_bf16(a[ms], b, acc[ms][j]);
        }
    }
    // C/D layout: VGPR r -> M = r (lanes 0-15) / M = 8+r (lanes 16-31), N = lane%16
    int colLane = lane & 15, hi = lane >> 4;
#pragma unroll
    for (int j = 0; j < 4; ++j) {
        int n = n0w + j * 16 + colLane;
        if (n < N) {
            float bb = bias ? bias[n] : 0.f;
#pragma unroll
            for (int ms = 0; ms < 4; ++ms) {
#pragma unroll
                for (int r = 0; r < 8; ++r) {
                    int m = mBase + ms * 16 + hi * 8 + r;
                    C[(size_t)m * Nstride + n] = acc[ms][j][r] + bb;
                }
            }
        }
    }
}

// ---------------------------------------------------------------------------
// Persistent bidirectional LSTM-with-projection recurrence.
// grid = 2 (blockIdx.x = direction), block = 1024 (32 waves, one WGP).
// Per step: gates = h_pad(16x512) @ WThh(->2048) via WMMA,
//           elementwise i/f/g/o + c update, h_raw = o*tanh(c),
//           h = h_raw_pad(16x512) @ WTproj(->512) via WMMA.
// Rows 4..15 of padded activations stay zero; batch lives in rows 0..3.
// ---------------------------------------------------------------------------
__global__ __launch_bounds__(1024, 1)
void lstm_dir_kernel(const float* __restrict__ XPf, const float* __restrict__ XPb,
                     const __bf16* __restrict__ WThh_f, const __bf16* __restrict__ WThh_b,
                     const __bf16* __restrict__ WTpj_f, const __bf16* __restrict__ WTpj_b,
                     const float* __restrict__ bias_f, const float* __restrict__ bias_b,
                     __bf16* __restrict__ Hcat) {
    int dir = blockIdx.x;
    const float*  XP   = dir ? XPb    : XPf;
    const __bf16* WThh = dir ? WThh_b : WThh_f;
    const __bf16* WTpj = dir ? WTpj_b : WTpj_f;
    const float*  bias = dir ? bias_b : bias_f;

    __shared__ __bf16 hpad[16 * HH];     // 16 KB  (bf16 h, padded to 16 rows)
    __shared__ float  gates[4 * G4H];    // 32 KB  (h @ W_hh)
    __shared__ __bf16 rpad[16 * HH];     // 16 KB  (bf16 o*tanh(c), padded)

    int tid = threadIdx.x, wave = tid >> 5, lane = tid & 31;

    for (int i = tid; i < 16 * HH; i += 1024) {
        hpad[i] = (__bf16)0.f;
        rpad[i] = (__bf16)0.f;
    }
    float c0 = 0.f, c1 = 0.f;            // each thread owns 2 cell elements
    int e0 = tid, e1 = tid + 1024;
    int b0 = e0 >> 9, j0 = e0 & 511;
    int b1 = e1 >> 9, j1 = e1 & 511;
    __syncthreads();

    for (int s = 0; s < TT; ++s) {
        int t = dir ? (TT - 1 - s) : s;

        // ---- phase A: gates = hpad @ WThh  (16x512 * 512x2048), wave -> 64 cols
        {
            int n0w = wave * 64;
            v8f acc[4] = {};
            for (int k0 = 0; k0 < HH; k0 += 32) {
                v16bf a = frag_A(hpad, HH, k0, lane);
#pragma unroll
                for (int j = 0; j < 4; ++j) {
                    v16bf b = frag_B(WThh, HH, n0w + j * 16, k0, lane);
                    acc[j] = wmma_bf16(a, b, acc[j]);
                }
            }
            if (lane < 16) {
#pragma unroll
                for (int j = 0; j < 4; ++j) {
                    int n = n0w + j * 16 + lane;
#pragma unroll
                    for (int r = 0; r < 4; ++r)       // rows 0..3 = batch
                        gates[r * G4H + n] = acc[j][r];
                }
            }
        }
        __syncthreads();

        // ---- phase B: elementwise gate nonlinearities + cell update
        {
            {
                const float* xp = XP + ((size_t)(b0 * TT + t)) * G4H;
                const float* g  = gates + b0 * G4H;
                float pi = g[j0]            + xp[j0]            + bias[j0];
                float pf = g[512 + j0]      + xp[512 + j0]      + bias[512 + j0];
                float pg = g[1024 + j0]     + xp[1024 + j0]     + bias[1024 + j0];
                float po = g[1536 + j0]     + xp[1536 + j0]     + bias[1536 + j0];
                float ig = 1.f / (1.f + expf(-pi));
                float fg = 1.f / (1.f + expf(-pf));
                float og = 1.f / (1.f + expf(-po));
                c0 = ig * tanhf(pg) + fg * c0;
                rpad[b0 * HH + j0] = (__bf16)(og * tanhf(c0));
            }
            {
                const float* xp = XP + ((size_t)(b1 * TT + t)) * G4H;
                const float* g  = gates + b1 * G4H;
                float pi = g[j1]            + xp[j1]            + bias[j1];
                float pf = g[512 + j1]      + xp[512 + j1]      + bias[512 + j1];
                float pg = g[1024 + j1]     + xp[1024 + j1]     + bias[1024 + j1];
                float po = g[1536 + j1]     + xp[1536 + j1]     + bias[1536 + j1];
                float ig = 1.f / (1.f + expf(-pi));
                float fg = 1.f / (1.f + expf(-pf));
                float og = 1.f / (1.f + expf(-po));
                c1 = ig * tanhf(pg) + fg * c1;
                rpad[b1 * HH + j1] = (__bf16)(og * tanhf(c1));
            }
        }
        __syncthreads();

        // ---- phase C: h = rpad @ WTproj (16x512 * 512x512), wave -> 16 cols
        {
            int n0 = wave * 16;
            v8f acc = {};
            for (int k0 = 0; k0 < HH; k0 += 32) {
                v16bf a = frag_A(rpad, HH, k0, lane);
                v16bf b = frag_B(WTpj, HH, n0, k0, lane);
                acc = wmma_bf16(a, b, acc);
            }
            if (lane < 16) {
#pragma unroll
                for (int r = 0; r < 4; ++r) {
                    __bf16 hv = (__bf16)acc[r];
                    hpad[r * HH + n0 + lane] = hv;   // state for next step
                    Hcat[((size_t)(dir * MROWS + r * TT + t)) * HH + n0 + lane] = hv;
                }
            }
        }
        __syncthreads();
    }
}

// ---------------------------------------------------------------------------
// Host-side orchestration
// ---------------------------------------------------------------------------
extern "C" void kernel_launch(void* const* d_in, const int* in_sizes, int n_in,
                              void* d_out, int out_size, void* d_ws, size_t ws_size,
                              hipStream_t stream) {
    const int*   tokens = (const int*)d_in[0];
    const float* emb    = (const float*)d_in[1];
    const float* Wih_f  = (const float*)d_in[2];
    const float* Whh_f  = (const float*)d_in[3];
    const float* b_f    = (const float*)d_in[4];
    const float* Wpj_f  = (const float*)d_in[5];
    const float* Wih_b  = (const float*)d_in[6];
    const float* Whh_b  = (const float*)d_in[7];
    const float* b_b    = (const float*)d_in[8];
    const float* Wpj_b  = (const float*)d_in[9];
    const float* Wlin   = (const float*)d_in[10];
    const float* blin   = (const float*)d_in[11];
    float* out = (float*)d_out;

    char* ws = (char*)d_ws;
    size_t off = 0;
    auto wsAlloc = [&](size_t bytes) -> void* {
        void* p = (void*)(ws + off);
        off += (bytes + 255) & ~(size_t)255;
        return p;
    };

    __bf16* Aemb   = (__bf16*)wsAlloc((size_t)MROWS * EPAD * 2);
    __bf16* WTih_f = (__bf16*)wsAlloc((size_t)G4H * EPAD * 2);
    __bf16* WTih_b = (__bf16*)wsAlloc((size_t)G4H * EPAD * 2);
    __bf16* WThh_f = (__bf16*)wsAlloc((size_t)G4H * HH * 2);
    __bf16* WThh_b = (__bf16*)wsAlloc((size_t)G4H * HH * 2);
    __bf16* WTpj_f = (__bf16*)wsAlloc((size_t)HH * HH * 2);
    __bf16* WTpj_b = (__bf16*)wsAlloc((size_t)HH * HH * 2);
    __bf16* WTlin  = (__bf16*)wsAlloc((size_t)NLIN_PAD * HH * 2);
    float*  XPf    = (float*)wsAlloc((size_t)MROWS * G4H * 4);
    float*  XPb    = (float*)wsAlloc((size_t)MROWS * G4H * 4);
    __bf16* Hcat   = (__bf16*)wsAlloc((size_t)2 * MROWS * HH * 2);

    // 1) embedding gather -> bf16 padded activations
    gather_emb_kernel<<<MROWS, 256, 0, stream>>>(tokens, emb, Aemb);

    // 2) weight conversions (f32 [K][N] -> bf16 transposed [Np][Kp])
    transpose_bf16_kernel<<<dim3(EPAD / 16, G4H / 16), 256, 0, stream>>>(Wih_f, WTih_f, EE, G4H, EPAD, G4H);
    transpose_bf16_kernel<<<dim3(EPAD / 16, G4H / 16), 256, 0, stream>>>(Wih_b, WTih_b, EE, G4H, EPAD, G4H);
    transpose_bf16_kernel<<<dim3(HH / 16, G4H / 16), 256, 0, stream>>>(Whh_f, WThh_f, HH, G4H, HH, G4H);
    transpose_bf16_kernel<<<dim3(HH / 16, G4H / 16), 256, 0, stream>>>(Whh_b, WThh_b, HH, G4H, HH, G4H);
    transpose_bf16_kernel<<<dim3(HH / 16, HH / 16), 256, 0, stream>>>(Wpj_f, WTpj_f, HH, HH, HH, HH);
    transpose_bf16_kernel<<<dim3(HH / 16, HH / 16), 256, 0, stream>>>(Wpj_b, WTpj_b, HH, HH, HH, HH);
    transpose_bf16_kernel<<<dim3(HH / 16, NLIN_PAD / 16), 256, 0, stream>>>(Wlin, WTlin, HH, VV, HH, NLIN_PAD);

    // 3) input projections xp = x @ W_ih  ([2048,320]@[320,2048] -> f32)
    gemm_bf16_kernel<<<dim3(G4H / 512, MROWS / 64), 256, 0, stream>>>(Aemb, WTih_f, XPf, nullptr, G4H, G4H, EPAD);
    gemm_bf16_kernel<<<dim3(G4H / 512, MROWS / 64), 256, 0, stream>>>(Aemb, WTih_b, XPb, nullptr, G4H, G4H, EPAD);

    // 4) both LSTM directions (persistent workgroups, sequential over T)
    lstm_dir_kernel<<<2, 1024, 0, stream>>>(XPf, XPb, WThh_f, WThh_b,
                                            WTpj_f, WTpj_b, b_f, b_b, Hcat);

    // 5) vocab projection: [4096,512] @ [512,50257] + b_lin -> d_out (HBM-bound)
    gemm_bf16_kernel<<<dim3(NLIN_PAD / 512, 2 * MROWS / 64), 256, 0, stream>>>(
        Hcat, WTlin, out, blin, VV, VV, HH);
}
